// GraphTransformerDecode_s2_67568425501199
// MI455X (gfx1250) — compile-verified, matrix operand order checked by reading, not run on hardware
//
#include <hip/hip_runtime.h>
#include <hip/hip_bf16.h>

// MI455X / gfx1250, wave32, WMMA f16 16x16x32 path.

typedef __attribute__((ext_vector_type(16))) _Float16 v16h;
typedef __attribute__((ext_vector_type(8)))  _Float16 v8h;
typedef __attribute__((ext_vector_type(8)))  float    v8f;

#define DD   512          // feature dim
#define LDU  520          // padded f16 LDS stride (1040B, 16B aligned, bank-spread)
#define LDF  516          // padded f32 LDS stride

// ---- WMMA fragment helpers -------------------------------------------------
// 16-bit A-matrix 16x32 layout (ISA 7.12.2): lane L<16 -> row=L, halves 0..7
// hold K=k0+0..7, halves 8..15 hold K=k0+16..23; lanes>=16 same row, K shifted
// by +8.  B (Wt pre-transposed to [n][k]) uses the mirrored mapping with
// row := n.  Both reduce to two contiguous 16-byte loads per lane.
__device__ inline const _Float16* frag_addr(const _Float16* base, int ld, int k0, int lane) {
  return base + (lane & 15) * ld + k0 + ((lane >> 4) << 3);
}
__device__ inline v16h frag_ld(const _Float16* p) {
  v8h lo = *(const v8h*)(p);
  v8h hi = *(const v8h*)(p + 16);
  v16h f;
#pragma unroll
  for (int i = 0; i < 8; ++i) { f[i] = lo[i]; f[i + 8] = hi[i]; }
  return f;
}
__device__ inline v8f wmma(v16h a, v16h b, v8f c) {
  return __builtin_amdgcn_wmma_f32_16x16x32_f16(false, a, false, b, (short)0, c, false, false);
}
__device__ inline void wave_reduce2(float& s, float& ss) {
#pragma unroll
  for (int m = 16; m >= 1; m >>= 1) {
    s  += __shfl_xor(s,  m, 32);
    ss += __shfl_xor(ss, m, 32);
  }
}
// Prefetch the wave's 64-row (64KB) B-panel: 16 instrs x 32 lanes x 128B lines.
__device__ inline void panel_prefetch(const _Float16* Bpanel, int lane) {
#pragma unroll
  for (int j = 0; j < 16; ++j)
    __builtin_prefetch((const char*)Bpanel + (j * 32 + lane) * 128, 0, 1);
}

// ---- software-pipelined 16xK GEMM core (A: global or LDS, B: 4 N-tiles) ----
// Loads for step k are issued before the WMMAs consuming step k-1, so the
// scheduler can use partial s_wait_loadcnt and overlap VMEM with the XDL burst.
__device__ inline void gemm_core(const _Float16* Abase, int lda,
                                 const _Float16* Bp, v8f* acc, int lane) {
  v16h a  = frag_ld(frag_addr(Abase, lda, 0, lane));
  v16h b0 = frag_ld(frag_addr(Bp + 0 * 16 * DD, DD, 0, lane));
  v16h b1 = frag_ld(frag_addr(Bp + 1 * 16 * DD, DD, 0, lane));
  v16h b2 = frag_ld(frag_addr(Bp + 2 * 16 * DD, DD, 0, lane));
  v16h b3 = frag_ld(frag_addr(Bp + 3 * 16 * DD, DD, 0, lane));
#pragma unroll 2
  for (int k0 = 32; k0 < DD; k0 += 32) {
    v16h an = frag_ld(frag_addr(Abase, lda, k0, lane));
    v16h c0 = frag_ld(frag_addr(Bp + 0 * 16 * DD, DD, k0, lane));
    v16h c1 = frag_ld(frag_addr(Bp + 1 * 16 * DD, DD, k0, lane));
    v16h c2 = frag_ld(frag_addr(Bp + 2 * 16 * DD, DD, k0, lane));
    v16h c3 = frag_ld(frag_addr(Bp + 3 * 16 * DD, DD, k0, lane));
    acc[0] = wmma(a, b0, acc[0]);
    acc[1] = wmma(a, b1, acc[1]);
    acc[2] = wmma(a, b2, acc[2]);
    acc[3] = wmma(a, b3, acc[3]);
    a = an; b0 = c0; b1 = c1; b2 = c2; b3 = c3;
  }
  acc[0] = wmma(a, b0, acc[0]);
  acc[1] = wmma(a, b1, acc[1]);
  acc[2] = wmma(a, b2, acc[2]);
  acc[3] = wmma(a, b3, acc[3]);
}

// ---- weight prep -----------------------------------------------------------
__global__ __launch_bounds__(256) void k_w2h_t(const float* __restrict__ W,
                                               _Float16* __restrict__ Wt) {
  int idx = blockIdx.x * 256 + threadIdx.x;      // 512*512
  int n = idx >> 9, k = idx & 511;
  Wt[idx] = (_Float16)W[k * DD + n];             // transpose: Wt[n][k] = W[k][n]
}
__global__ __launch_bounds__(256) void k_f2h(const float* __restrict__ X,
                                             _Float16* __restrict__ Xh, int n) {
  int idx = blockIdx.x * 256 + threadIdx.x;
  if (idx < n) Xh[idx] = (_Float16)X[idx];
}

// ---- generic WMMA GEMM: C[M x 512] = A[M x 512] @ Wt^T (+bias) -------------
__global__ __launch_bounds__(256) void k_gemm(const _Float16* __restrict__ A,
                                              const _Float16* __restrict__ Wt,
                                              const float* __restrict__ bias,
                                              float* __restrict__ C, int hasBias) {
  const int lane = threadIdx.x & 31;
  const int wv   = threadIdx.x >> 5;
  const int r0   = blockIdx.x * 16;
  const _Float16* Abase = A + r0 * DD;
  const _Float16* Bp    = Wt + (wv * 64) * DD;
  panel_prefetch(Bp, lane);
  v8f acc[4] = {};
  gemm_core(Abase, DD, Bp, acc, lane);
#pragma unroll
  for (int t = 0; t < 4; ++t) {
    int col = wv * 64 + t * 16 + (lane & 15);
    float bv = hasBias ? bias[col] : 0.f;
#pragma unroll
    for (int v = 0; v < 8; ++v) {
      int row = v + ((lane >> 4) << 3);            // C layout: VGPR v -> M=v(+8)
      C[(r0 + row) * DD + col] = acc[t][v] + bv;
    }
  }
}

// ---- MAB1 pre: T = LN(S1[q]*A1[b] + bq1 + V1[b]) ---------------------------
__global__ __launch_bounds__(256) void k_mab1pre(const float* __restrict__ A1,
                                                 const float* __restrict__ V1,
                                                 const float* __restrict__ S1,
                                                 const float* __restrict__ bq1,
                                                 const float* __restrict__ g,
                                                 const float* __restrict__ bt,
                                                 float* __restrict__ Tf,
                                                 _Float16* __restrict__ Th) {
  const int lane = threadIdx.x & 31;
  const int r = blockIdx.x * 8 + (threadIdx.x >> 5);   // row in [0, 2048)
  const int b = r >> 1, q = r & 1;
  const float sc = S1[q];
  float s = 0.f, ss = 0.f, vals[16];
#pragma unroll
  for (int j = 0; j < 16; ++j) {
    int d = lane + 32 * j;
    float x = sc * A1[b * DD + d] + bq1[d] + V1[b * DD + d];
    vals[j] = x; s += x; ss += x * x;
  }
  wave_reduce2(s, ss);
  float mean = s * (1.f / DD);
  float inv  = rsqrtf(ss * (1.f / DD) - mean * mean + 1e-5f);
#pragma unroll
  for (int j = 0; j < 16; ++j) {
    int d = lane + 32 * j;
    float y = g[d] * (vals[j] - mean) * inv + bt[d];
    Tf[r * DD + d] = y; Th[r * DD + d] = (_Float16)y;
  }
}

// ---- fused GEMM + (relu, residual, LN) epilogue:  out = LN(R + relu(A@W+b))-
__global__ __launch_bounds__(256) void k_gemm_lnepi(const _Float16* __restrict__ Ah,
                                                    const float* __restrict__ Resid,
                                                    const _Float16* __restrict__ Wt,
                                                    const float* __restrict__ bias,
                                                    const float* __restrict__ g,
                                                    const float* __restrict__ bt,
                                                    float* __restrict__ outF,
                                                    _Float16* __restrict__ outH) {
  __shared__ __align__(16) float Z[16][LDF];
  const int lane = threadIdx.x & 31;
  const int wv   = threadIdx.x >> 5;
  const int r0   = blockIdx.x * 16;
  const _Float16* Abase = Ah + r0 * DD;
  const _Float16* Bp    = Wt + (wv * 64) * DD;
  panel_prefetch(Bp, lane);
  v8f acc[4] = {};
  gemm_core(Abase, DD, Bp, acc, lane);
#pragma unroll
  for (int t = 0; t < 4; ++t) {
    int col = wv * 64 + t * 16 + (lane & 15);
#pragma unroll
    for (int v = 0; v < 8; ++v) {
      int row = v + ((lane >> 4) << 3);
      float z = fmaxf(acc[t][v] + bias[col], 0.f) + Resid[(r0 + row) * DD + col];
      Z[row][col] = z;
    }
  }
  __syncthreads();
#pragma unroll
  for (int i = 0; i < 2; ++i) {                      // wave handles 2 rows
    int row = wv * 2 + i;
    float s = 0.f, ss = 0.f, vals[16];
#pragma unroll
    for (int j = 0; j < 16; ++j) {
      float x = Z[row][lane + 32 * j]; vals[j] = x; s += x; ss += x * x;
    }
    wave_reduce2(s, ss);
    float mean = s * (1.f / DD);
    float inv  = rsqrtf(ss * (1.f / DD) - mean * mean + 1e-5f);
#pragma unroll
    for (int j = 0; j < 16; ++j) {
      int d = lane + 32 * j;
      float y = g[d] * (vals[j] - mean) * inv + bt[d];
      outF[(r0 + row) * DD + d] = y; outH[(r0 + row) * DD + d] = (_Float16)y;
    }
  }
}

// ---- sl row-pair sum (key aggregate) ---------------------------------------
__global__ __launch_bounds__(256) void k_sum(const float* __restrict__ sl,
                                             _Float16* __restrict__ sumh) {
  int idx = blockIdx.x * 256 + threadIdx.x;          // 1024*512
  int b = idx >> 9, d = idx & 511;
  sumh[idx] = (_Float16)(sl[(2 * b) * DD + d] + sl[(2 * b + 1) * DD + d]);
}

// ---- final fused MAB2 kernel ----------------------------------------------
// rows r = b*64+n ; 16 rows per WG (single b per WG since 16 | 64).
// U = LN(S2[n,0]*sq0 + S2[n,1]*sq1 + bq2 + v2[b]); out = LN(U + relu(U@Wo2+bo2))
__global__ __launch_bounds__(256) void k_final(const float* __restrict__ sq,
                                               const float* __restrict__ v2,
                                               const float* __restrict__ S2,
                                               const float* __restrict__ bq2,
                                               const float* __restrict__ g2a,
                                               const float* __restrict__ b2a,
                                               const _Float16* __restrict__ Wo2t,
                                               const float* __restrict__ bo2,
                                               const float* __restrict__ g2b,
                                               const float* __restrict__ b2b,
                                               float* __restrict__ out) {
  __shared__ __align__(16) _Float16 Uh[16][LDU];
  __shared__ __align__(16) float    Uf[16][LDF];
  const int lane = threadIdx.x & 31;
  const int wv   = threadIdx.x >> 5;
  const int r0   = blockIdx.x * 16;
  const int b    = r0 >> 6;
  const int nb   = r0 & 63;
  const float* sq0 = sq + (2 * b) * DD;
  const float* sq1 = sq0 + DD;
  const float* v2r = v2 + b * DD;
  const _Float16* Bp = Wo2t + (wv * 64) * DD;
  panel_prefetch(Bp, lane);

  // ---- prologue: build U rows in LDS (f32 + f16) ----
#pragma unroll
  for (int i = 0; i < 2; ++i) {
    int row = wv * 2 + i;
    int n   = nb + row;
    float s0 = S2[n * 2 + 0], s1 = S2[n * 2 + 1];
    float s = 0.f, ss = 0.f, vals[16];
#pragma unroll
    for (int j = 0; j < 16; ++j) {
      int d = lane + 32 * j;
      float x = s0 * sq0[d] + s1 * sq1[d] + bq2[d] + v2r[d];
      vals[j] = x; s += x; ss += x * x;
    }
    wave_reduce2(s, ss);
    float mean = s * (1.f / DD);
    float inv  = rsqrtf(ss * (1.f / DD) - mean * mean + 1e-5f);
#pragma unroll
    for (int j = 0; j < 16; ++j) {
      int d = lane + 32 * j;
      float u = g2a[d] * (vals[j] - mean) * inv + b2a[d];
      Uf[row][d] = u; Uh[row][d] = (_Float16)u;
    }
  }
  __syncthreads();

  // ---- WMMA FFN GEMM: acc = U @ Wo2 (A from LDS, B streamed from L2) ----
  v8f acc[4] = {};
  gemm_core(&Uh[0][0], LDU, Bp, acc, lane);

  // ---- epilogue: z = relu(acc+bo2) + U  (same-thread read/modify of Uf) ----
#pragma unroll
  for (int t = 0; t < 4; ++t) {
    int col = wv * 64 + t * 16 + (lane & 15);
#pragma unroll
    for (int v = 0; v < 8; ++v) {
      int row = v + ((lane >> 4) << 3);
      float z = fmaxf(acc[t][v] + bo2[col], 0.f) + Uf[row][col];
      Uf[row][col] = z;
    }
  }
  __syncthreads();

  // ---- final LN + store ----
#pragma unroll
  for (int i = 0; i < 2; ++i) {
    int row = wv * 2 + i;
    float s = 0.f, ss = 0.f, vals[16];
#pragma unroll
    for (int j = 0; j < 16; ++j) {
      float x = Uf[row][lane + 32 * j]; vals[j] = x; s += x; ss += x * x;
    }
    wave_reduce2(s, ss);
    float mean = s * (1.f / DD);
    float inv  = rsqrtf(ss * (1.f / DD) - mean * mean + 1e-5f);
#pragma unroll
    for (int j = 0; j < 16; ++j) {
      int d = lane + 32 * j;
      out[(r0 + row) * DD + d] = g2b[d] * (vals[j] - mean) * inv + b2b[d];
    }
  }
}

// ---------------------------------------------------------------------------
extern "C" void kernel_launch(void* const* d_in, const int* in_sizes, int n_in,
                              void* d_out, int out_size, void* d_ws, size_t ws_size,
                              hipStream_t stream) {
  const int B = 1024;
  const float* X   = (const float*)d_in[0];
  const float* S1  = (const float*)d_in[1];
  const float* S2  = (const float*)d_in[2];
  const float* Wq1 = (const float*)d_in[3];  const float* bq1 = (const float*)d_in[4];
  const float* Wv1 = (const float*)d_in[7];  const float* bv1 = (const float*)d_in[8];
  const float* Wo1 = (const float*)d_in[9];  const float* bo1 = (const float*)d_in[10];
  const float* g1a = (const float*)d_in[11]; const float* b1a = (const float*)d_in[12];
  const float* g1b = (const float*)d_in[13]; const float* b1b = (const float*)d_in[14];
  const float* Wq2 = (const float*)d_in[15]; const float* bq2 = (const float*)d_in[16];
  const float* Wv2 = (const float*)d_in[19]; const float* bv2 = (const float*)d_in[20];
  const float* Wo2 = (const float*)d_in[21]; const float* bo2 = (const float*)d_in[22];
  const float* g2a = (const float*)d_in[23]; const float* b2a = (const float*)d_in[24];
  const float* g2b = (const float*)d_in[25]; const float* b2b = (const float*)d_in[26];
  // Wk1/Wk2 (d_in[5],[17]) are provably dead: softmax over 1 key / 2 identical keys.

  // ---- workspace carve-up (~28.3 MB) ----
  const size_t WSZ = (size_t)DD * DD;          // 262144
  _Float16* p16 = (_Float16*)d_ws;
  _Float16* Wq1t = p16;            _Float16* Wv1t = Wq1t + WSZ;
  _Float16* Wo1t = Wv1t + WSZ;     _Float16* Wq2t = Wo1t + WSZ;
  _Float16* Wv2t = Wq2t + WSZ;     _Float16* Wo2t = Wv2t + WSZ;
  _Float16* Xh   = Wo2t + WSZ;                              // 1024*512 h
  float* f32base = (float*)(Xh + (size_t)B * DD);
  float* A1  = f32base;            float* V1  = A1 + (size_t)B * DD;
  float* Tf  = V1 + (size_t)B * DD;                         // 2048*512 f
  _Float16* Th = (_Float16*)(Tf + (size_t)2 * B * DD);      // 2048*512 h
  float* sl  = (float*)(Th + (size_t)2 * B * DD);           // 2048*512 f
  _Float16* slh = (_Float16*)(sl + (size_t)2 * B * DD);     // 2048*512 h
  float* sq  = (float*)(slh + (size_t)2 * B * DD);          // 2048*512 f
  _Float16* sumh = (_Float16*)(sq + (size_t)2 * B * DD);    // 1024*512 h
  float* v2r = (float*)(sumh + (size_t)B * DD);             // 1024*512 f

  const int WG = 1024;                                       // 512*512/256
  k_w2h_t<<<WG, 256, 0, stream>>>(Wq1, Wq1t);
  k_w2h_t<<<WG, 256, 0, stream>>>(Wv1, Wv1t);
  k_w2h_t<<<WG, 256, 0, stream>>>(Wo1, Wo1t);
  k_w2h_t<<<WG, 256, 0, stream>>>(Wq2, Wq2t);
  k_w2h_t<<<WG, 256, 0, stream>>>(Wv2, Wv2t);
  k_w2h_t<<<WG, 256, 0, stream>>>(Wo2, Wo2t);
  k_f2h<<<(B * DD + 255) / 256, 256, 0, stream>>>(X, Xh, B * DD);

  k_gemm<<<B / 16, 256, 0, stream>>>(Xh, Wq1t, nullptr, A1, 0);       // A1 = X@Wq1
  k_gemm<<<B / 16, 256, 0, stream>>>(Xh, Wv1t, bv1, V1, 1);           // V1 = X@Wv1+bv1
  k_mab1pre<<<2 * B / 8, 256, 0, stream>>>(A1, V1, S1, bq1, g1a, b1a, Tf, Th);
  k_gemm_lnepi<<<2 * B / 16, 256, 0, stream>>>(Th, Tf, Wo1t, bo1, g1b, b1b, sl, slh);
  k_gemm<<<2 * B / 16, 256, 0, stream>>>(slh, Wq2t, nullptr, sq, 0);  // sq = sl@Wq2
  k_sum<<<B * DD / 256, 256, 0, stream>>>(sl, sumh);
  k_gemm<<<B / 16, 256, 0, stream>>>(sumh, Wv2t, bv2, v2r, 1);        // v2 = sum@Wv2+bv2
  k_final<<<(B * 64) / 16, 256, 0, stream>>>(sq, v2r, S2, bq2, g2a, b2a,
                                             Wo2t, bo2, g2b, b2b, (float*)d_out);
}
